// LaserModel_77524159692849
// MI455X (gfx1250) — matrix operand order repaired and checked by reading, not compiled, not run on hardware
//
#include <hip/hip_runtime.h>
#include <hip/hip_bf16.h>

// SimpleRNN(relu) fused kernel for MI455X (gfx1250), wave32 + WMMA.
//   B=2048, T=1024, F=32, H=120
// One workgroup = 256 threads = 8 waves, owns 16 batch rows for the whole scan.
// Per step: A = [h_t (pad 128) | x_t (32)]  (K=160 = 5 chunks of 32),
//           W = [Wh; 0; Wx] in LDS f16, B-fragments register-resident.
// 5x v_wmma_f32_16x16x32_f16 per wave per step, h double-buffered in LDS.

typedef __attribute__((ext_vector_type(16))) _Float16 v16h;
typedef __attribute__((ext_vector_type(8)))  _Float16 h8;
typedef __attribute__((ext_vector_type(8)))  float    v8f;

#define B_   2048
#define T_   1024
#define F_   32
#define H_   120

#define AS   168      // A-tile row stride in halves (mult of 8 -> 16B aligned segs)
#define WS   128      // weight row stride in halves (N padded 120 -> 128)
#define KC   5        // K chunks in recurrence: 160 / 32

static __device__ __forceinline__ v16h make16(h8 lo, h8 hi) {
  return __builtin_shufflevector(lo, hi, 0,1,2,3,4,5,6,7,8,9,10,11,12,13,14,15);
}

// A fragment (16x32 f16), ISA 7.12.2: lane l -> row m = l&15,
// half e -> K = kbase + ((e&8)<<1) + ((l>>4)<<3) + (e&7)  => two contiguous 16B reads.
static __device__ __forceinline__ v16h load_a_frag(const _Float16* A, int lane, int kbase) {
  const int m  = lane & 15;
  const int hi = (lane >> 4) & 1;
  const _Float16* p = A + m * AS + kbase + hi * 8;
  h8 lo  = *(const h8*)(p);
  h8 hi8 = *(const h8*)(p + 16);
  return make16(lo, hi8);
}

// B fragment (32x16 f16): lane l -> column N = ncol + (l&15),
// half e -> K = kbase + ((e&8)<<1) + ((l>>4)<<3) + (e&7)  (same K order as A).
static __device__ __forceinline__ v16h load_b_frag(const _Float16* W, int lane, int kbase, int ncol) {
  const _Float16* p = W + (size_t)(kbase + ((lane >> 4) & 1) * 8) * WS + ncol + (lane & 15);
  v16h r;
#pragma unroll
  for (int e = 0; e < 16; ++e) r[e] = p[(size_t)(((e & 8) << 1) + (e & 7)) * WS];
  return r;
}

static __device__ __forceinline__ v8f wmma16(v16h a, v16h b, v8f c) {
  // (neg_a, A, neg_b, B, c_mod, C, reuse_a, reuse_b)
  return __builtin_amdgcn_wmma_f32_16x16x32_f16(false, a, false, b, (short)0, c, false, false);
}

__global__ __launch_bounds__(256) void LaserModel_rnn_kernel(
    const float* __restrict__ x,  const float* __restrict__ Wx,
    const float* __restrict__ Wh, const float* __restrict__ b,
    const float* __restrict__ W1, const float* __restrict__ b1,
    const float* __restrict__ W2, const float* __restrict__ b2,
    float* __restrict__ out)
{
  __shared__ __align__(16) _Float16 Wl[160 * WS];     // 40 KB: [Wh(120); pad(8); Wx(32)] x Npad128
  __shared__ __align__(16) _Float16 Ab[2][16 * AS];   // double-buffered A tile, ~10.5 KB
  __shared__ float W2l[H_];

  const int tid     = threadIdx.x;
  const int lane    = tid & 31;
  const int wv      = tid >> 5;            // 0..7 -> N-tile
  const int rowbase = blockIdx.x * 16;     // 16 batch rows per WG

  // ---- stage weights: Wcomb = [Wh ; 0 ; Wx], zero-padded to [160,128] f16
  for (int i = tid; i < 160 * WS; i += 256) {
    const int r = i >> 7, n = i & 127;
    float v = 0.f;
    if (n < H_) {
      if (r < H_)        v = Wh[r * H_ + n];
      else if (r >= 128) v = Wx[(r - 128) * H_ + n];
    }
    Wl[i] = (_Float16)v;
  }
  // ---- A buffer 0: h0 = 0 everywhere, then x_0 into columns 128..159
  for (int i = tid; i < 16 * AS; i += 256) Ab[0][i] = (_Float16)0.f;
  for (int i = tid; i < 16 * F_; i += 256) {
    const int m = i >> 5, f = i & 31;
    Ab[0][m * AS + 128 + f] = (_Float16)x[(size_t)(rowbase + m) * (T_ * F_) + f];
  }
  for (int i = tid; i < H_; i += 256) W2l[i] = W2[i];
  __syncthreads();

  // ---- loop-invariant B fragments (register resident) + bias
  const int ncol = wv * 16;
  const int nIdx = ncol + (lane & 15);     // this lane's output column N
  const float biasN = (nIdx < H_) ? b[nIdx] : 0.f;
  v16h bf[KC];
#pragma unroll
  for (int c = 0; c < KC; ++c) bf[c] = load_b_frag(Wl, lane, c * 32, ncol);

  const int hiHalf = (lane >> 4) & 1;
  const int m0 = tid >> 5,          f0 = tid & 31;           // x staging: 512 elems / 256 thr
  const int m1 = (tid + 256) >> 5,  f1 = tid & 31;

  // ---- the scan: h_{t+1} = relu([h_t | x_t] @ [Wh;Wx] + b)
#pragma unroll 2
  for (int t = 0; t < T_; ++t) {
    const _Float16* Ac = Ab[t & 1];
    _Float16*       An = Ab[(t & 1) ^ 1];

    // issue next-step x loads early (latency hiding)
    float xv0 = 0.f, xv1 = 0.f;
    if (t + 1 < T_) {
      const size_t toff = (size_t)(t + 1) * F_;
      xv0 = x[(size_t)(rowbase + m0) * (T_ * F_) + toff + f0];
      xv1 = x[(size_t)(rowbase + m1) * (T_ * F_) + toff + f1];
    }

    v8f acc0, acc1 = {};
#pragma unroll
    for (int r = 0; r < 8; ++r) acc0[r] = biasN;

    // 5 K-chunks, two independent accumulation chains for ILP
    acc0 = wmma16(load_a_frag(Ac, lane,   0), bf[0], acc0);
    acc1 = wmma16(load_a_frag(Ac, lane,  32), bf[1], acc1);
    acc0 = wmma16(load_a_frag(Ac, lane,  64), bf[2], acc0);
    acc1 = wmma16(load_a_frag(Ac, lane,  96), bf[3], acc1);
    acc0 = wmma16(load_a_frag(Ac, lane, 128), bf[4], acc0);

    if (t + 1 < T_) {
      An[m0 * AS + 128 + f0] = (_Float16)xv0;
      An[m1 * AS + 128 + f1] = (_Float16)xv1;
    }

    // relu + write new h tile (f16) into the other buffer; keep pad cols zero
#pragma unroll
    for (int r = 0; r < 8; ++r) {
      float v = acc0[r] + acc1[r];
      v = v > 0.f ? v : 0.f;
      An[(r + hiHalf * 8) * AS + nIdx] = (nIdx < H_) ? (_Float16)v : (_Float16)0.f;
    }
    __syncthreads();
  }
  // T even -> h_last lives (f16, pad cols zero) in Ab[0]

  // ---- dense head 1: relu(h_last @ W1 + b1), reuse weight LDS region
  for (int i = tid; i < 128 * WS; i += 256) {
    const int r = i >> 7, n = i & 127;
    Wl[i] = (_Float16)((r < H_ && n < H_) ? W1[r * H_ + n] : 0.f);
  }
  __syncthreads();

  const float bias1 = (nIdx < H_) ? b1[nIdx] : 0.f;
  v8f d0, d1 = {};
#pragma unroll
  for (int r = 0; r < 8; ++r) d0[r] = bias1;
  d0 = wmma16(load_a_frag(Ab[0], lane,  0), load_b_frag(Wl, lane,  0, ncol), d0);
  d1 = wmma16(load_a_frag(Ab[0], lane, 32), load_b_frag(Wl, lane, 32, ncol), d1);
  d0 = wmma16(load_a_frag(Ab[0], lane, 64), load_b_frag(Wl, lane, 64, ncol), d0);
  d1 = wmma16(load_a_frag(Ab[0], lane, 96), load_b_frag(Wl, lane, 96, ncol), d1);

#pragma unroll
  for (int r = 0; r < 8; ++r) {
    float v = d0[r] + d1[r];
    v = v > 0.f ? v : 0.f;
    Ab[1][(r + hiHalf * 8) * AS + nIdx] = (nIdx < H_) ? (_Float16)v : (_Float16)0.f;
  }
  __syncthreads();

  // ---- dense head 2: [16,120] @ [120,1] + b2
  if (tid < 16) {
    float s = b2[0];
    for (int k = 0; k < H_; ++k) s += (float)Ab[1][tid * AS + k] * W2l[k];
    out[rowbase + tid] = s;
  }
}

extern "C" void kernel_launch(void* const* d_in, const int* in_sizes, int n_in,
                              void* d_out, int out_size, void* d_ws, size_t ws_size,
                              hipStream_t stream) {
  (void)in_sizes; (void)n_in; (void)d_ws; (void)ws_size; (void)out_size;
  const float* x  = (const float*)d_in[0];
  const float* Wx = (const float*)d_in[1];
  const float* Wh = (const float*)d_in[2];
  const float* b  = (const float*)d_in[3];
  const float* W1 = (const float*)d_in[4];
  const float* b1 = (const float*)d_in[5];
  const float* W2 = (const float*)d_in[6];
  const float* b2 = (const float*)d_in[7];
  float* out = (float*)d_out;

  hipLaunchKernelGGL(LaserModel_rnn_kernel, dim3(B_ / 16), dim3(256), 0, stream,
                     x, Wx, Wh, b, W1, b1, W2, b2, out);
}
